// Attention_4088808866263
// MI455X (gfx1250) — matrix operand order repaired and checked by reading, not compiled, not run on hardware
//
#include <hip/hip_runtime.h>

#define B_  2
#define T_  2048
#define D_  2048
#define H_  16
#define HD_ 128

typedef __attribute__((ext_vector_type(16))) __bf16 v16bf;
typedef __attribute__((ext_vector_type(8)))  float  v8f;
typedef __attribute__((ext_vector_type(4)))  unsigned int v4u_;
typedef __attribute__((ext_vector_type(8)))  int v8i_;
typedef __attribute__((ext_vector_type(4)))  int v4i_;

#if __has_builtin(__builtin_amdgcn_tensor_load_to_lds) && __has_builtin(__builtin_amdgcn_s_wait_tensorcnt)
#define HAVE_TDM 1
#else
#define HAVE_TDM 0
#endif

union FragU { unsigned u[8]; v16bf v; };

__device__ __forceinline__ v8f zero8() {
  v8f z;
  #pragma unroll
  for (int i = 0; i < 8; i++) z[i] = 0.0f;
  return z;
}

// round-to-nearest-even fp32 -> bf16
__device__ __forceinline__ unsigned short f2bf(float f) {
  unsigned u = __float_as_uint(f);
  u += 0x7FFFu + ((u >> 16) & 1u);
  return (unsigned short)(u >> 16);
}
__device__ __forceinline__ unsigned pack2(float a, float b) {
  return (unsigned)f2bf(a) | ((unsigned)f2bf(b) << 16);
}

#if HAVE_TDM
// LDS byte offset of a __shared__ object (addrspace(3) addresses == DS offsets)
__device__ __forceinline__ unsigned lds_offset(const void* p) {
  return (unsigned)(unsigned long long)
      (__attribute__((address_space(3))) const void*)p;
}

// TDM 2D bf16 tile load: Global (row stride stride0 elements) -> LDS, contiguous
// rows with optional LDS padding (g1w0_extra carries pad_enable/pad_interval bits).
__device__ __forceinline__ void tdm_load_2d_bf16(
    unsigned lds_off, const void* gptr,
    unsigned tdim0, unsigned tdim1, unsigned long long stride0,
    unsigned tile0, unsigned tile1, unsigned g1w0_extra) {
  unsigned long long ga = (unsigned long long)(size_t)gptr;
  v4u_ g0;
  g0[0] = 1u;                                       // count=1, user descriptor
  g0[1] = lds_off;                                  // lds_addr (bytes)
  g0[2] = (unsigned)(ga & 0xFFFFFFFFu);             // global_addr[31:0]
  g0[3] = (unsigned)((ga >> 32) & 0x01FFFFFFu)      // global_addr[56:32]
        | (2u << 30);                               // type=2 ("image")
  v8i_ g1;
  g1[0] = (int)((1u << 16) | g1w0_extra);           // data_size=1 (2B) | pad bits
  g1[1] = (int)((tdim0 & 0xFFFFu) << 16);           // atomic_addr=0 | dim0[15:0]
  g1[2] = (int)((tdim0 >> 16) | ((tdim1 & 0xFFFFu) << 16));
  g1[3] = (int)((tdim1 >> 16) | (tile0 << 16));     // dim1[31:16] | tile_dim0
  g1[4] = (int)(tile1 & 0xFFFFu);                   // tile_dim1 | tile_dim2=0
  g1[5] = (int)(unsigned)(stride0 & 0xFFFFFFFFull); // dim0_stride[31:0]
  g1[6] = (int)(unsigned)((stride0 >> 32) & 0xFFFFull);
  g1[7] = 0;
  v4i_ g2 = {0, 0, 0, 0};
  v4i_ g3 = {0, 0, 0, 0};
#if defined(__clang_major__) && __clang_major__ >= 23
  v8i_ g4 = {0, 0, 0, 0, 0, 0, 0, 0};
  __builtin_amdgcn_tensor_load_to_lds(g0, g1, g2, g3, g4, 0);
#else
  __builtin_amdgcn_tensor_load_to_lds(g0, g1, g2, g3, 0);
#endif
}
#endif  // HAVE_TDM

// C[M,N] = A[M,K] * W[K,N] + bias[N]
// A: fp32 (A_BF16=false) or bf16 (A_BF16=true); C: bf16 or fp32
template <bool A_BF16, bool OUT_F32>
__global__ __launch_bounds__(256) void gemm_wmma(
    const void* __restrict__ Aptr, const float* __restrict__ W,
    const float* __restrict__ bias, void* __restrict__ Cptr,
    int M, int N, int K) {
  __shared__ unsigned As[128 * 17];   // [row][kpair] packed bf16 x2 along K
  __shared__ unsigned Bs[16 * 132];   // [kpair][n]  packed bf16 x2 along K
  const int tid  = threadIdx.x;
  const int lane = tid & 31;
  const int w    = tid >> 5;
  const int wm   = w >> 1, wn = w & 1;
  const int hl   = lane >> 4, l16 = lane & 15;
  const int blockRow = blockIdx.y * 128;
  const int blockCol = blockIdx.x * 128;

  v8f acc[2][4];
  #pragma unroll
  for (int i = 0; i < 2; i++)
    #pragma unroll
    for (int j = 0; j < 4; j++) acc[i][j] = zero8();

  const float*    Af  = (const float*)Aptr;
  const unsigned* A32 = (const unsigned*)Aptr;

#if HAVE_TDM
  const unsigned as_off = lds_offset(As);
#endif

  for (int kb = 0; kb < K; kb += 32) {
    __syncthreads();
    // ---- stage A tile 128x32 -> 128x16 packed pairs (LDS row stride 17 u32)
    if (A_BF16) {
#if HAVE_TDM
      if (w == 0) {
        // pad_enable=1 (bit20), pad_interval=3 -> 16 DWORDs (64B) per row,
        // pad_amount=0 -> 1 DWORD: hardware reproduces the stride-17 layout.
        const unsigned short* ga =
            (const unsigned short*)Aptr + (size_t)blockRow * K + kb;
        tdm_load_2d_bf16(as_off, ga, (unsigned)K, 1u << 20,
                         (unsigned long long)K, 32u, 128u,
                         (1u << 20) | (3u << 22));
      }
#else
      {
        int row = tid >> 1;
        int kp0 = (tid & 1) * 8;
        const unsigned* src =
            A32 + (size_t)(blockRow + row) * (size_t)(K >> 1) + (kb >> 1) + kp0;
        #pragma unroll
        for (int j = 0; j < 8; j++) As[row * 17 + kp0 + j] = src[j];
      }
#endif
    } else {
      int row = tid >> 1;
      int kp0 = (tid & 1) * 8;
      const float* src = Af + (size_t)(blockRow + row) * (size_t)K + kb + 2 * kp0;
      #pragma unroll
      for (int j = 0; j < 8; j++)
        As[row * 17 + kp0 + j] = pack2(src[2 * j], src[2 * j + 1]);
    }
    {   // ---- stage W tile 32x128 -> 16x128 packed pairs over K
      int kp = tid >> 4;
      int n0 = (tid & 15) * 8;
      const float* r0 = W + (size_t)(kb + 2 * kp) * N + blockCol + n0;
      const float* r1 = r0 + N;
      __builtin_prefetch(r0 + 32 * (size_t)N, 0, 0);   // next K-step of W
      #pragma unroll
      for (int j = 0; j < 8; j++) Bs[kp * 132 + n0 + j] = pack2(r0[j], r1[j]);
    }
#if HAVE_TDM
    if (A_BF16 && w == 0) __builtin_amdgcn_s_wait_tensorcnt(0);
#endif
    __syncthreads();

    FragU af[2];
    #pragma unroll
    for (int i = 0; i < 2; i++) {
      int row = wm * 32 + i * 16 + l16;
      #pragma unroll
      for (int vv = 0; vv < 8; vv++) {
        int kp2 = hl * 8 + vv;
        af[i].u[vv] = As[row * 17 + kp2];
      }
    }
    #pragma unroll
    for (int j = 0; j < 4; j++) {
      FragU bf;
      int n = wn * 64 + j * 16 + l16;
      #pragma unroll
      for (int vv = 0; vv < 8; vv++) bf.u[vv] = Bs[(hl * 8 + vv) * 132 + n];
      #pragma unroll
      for (int i = 0; i < 2; i++)
        acc[i][j] = __builtin_amdgcn_wmma_f32_16x16x32_bf16(
            false, af[i].v, false, bf.v, (short)0, acc[i][j], false, false);
    }
  }

  #pragma unroll
  for (int i = 0; i < 2; i++)
    #pragma unroll
    for (int j = 0; j < 4; j++) {
      int col = blockCol + wn * 64 + j * 16 + l16;
      float bval = bias[col];
      #pragma unroll
      for (int r = 0; r < 8; r++) {
        int row = blockRow + wm * 32 + i * 16 + r + 8 * hl;
        float val = acc[i][j][r] + bval;
        if (OUT_F32)
          ((float*)Cptr)[(size_t)row * N + col] = val;
        else
          ((unsigned short*)Cptr)[(size_t)row * N + col] = f2bf(val);
      }
    }
}

// Flash-attention with causal mask + ALiBi. 128 q-rows/block, 8 waves x 16 rows.
__global__ __launch_bounds__(256) void attn_wmma(
    const unsigned short* __restrict__ qbuf, const unsigned short* __restrict__ kbuf,
    const unsigned short* __restrict__ vbuf, const float* __restrict__ slopes,
    unsigned short* __restrict__ obuf) {
  __shared__ unsigned Ks[64 * 64];            // [key][dpair] (row-major K tile)
  __shared__ unsigned Vs[32 * 128];           // [keypair][d] packed over keys
  __shared__ unsigned short Ps[8][16 * 64];   // per-wave P tile
  const int tid = threadIdx.x;
  const int lane = tid & 31, w = tid >> 5;
  const int hl = lane >> 4, l16 = lane & 15;
  const int bh = blockIdx.y;
  const int b = bh >> 4, h = bh & 15;
  const int qBase = blockIdx.x * 128;
  const int qr0 = qBase + w * 16;
  const float slope = slopes[h];
  const float scale = 0.08838834764831845f;   // 1/sqrt(128)

  const unsigned* q32 = (const unsigned*)qbuf;
  const unsigned* v32 = (const unsigned*)vbuf;
#if !HAVE_TDM
  const unsigned* k32 = (const unsigned*)kbuf;
#else
  const unsigned ks_off = lds_offset(Ks);
#endif

  // preload Q fragments (16x128 => 4 k-steps of 16x32)
  FragU qf[4];
  {
    int row = qr0 + l16;
    size_t base = ((size_t)(b * T_ + row) * D_ + h * HD_) >> 1;
    #pragma unroll
    for (int ks = 0; ks < 4; ks++)
      #pragma unroll
      for (int vv = 0; vv < 8; vv++) {
        int kp2 = ks * 16 + hl * 8 + vv;
        qf[ks].u[vv] = q32[base + kp2];
      }
  }

  v8f o[8];
  #pragma unroll
  for (int n = 0; n < 8; n++) o[n] = zero8();
  float mrow[8], lsum[8];
  #pragma unroll
  for (int r = 0; r < 8; r++) { mrow[r] = -3.0e38f; lsum[r] = 0.0f; }

  const int nkt = blockIdx.x * 2 + 2;         // causal: keys <= qBase+127
  for (int kt = 0; kt < nkt; kt++) {
    const int keyBase = kt * 64;
    __syncthreads();
#if HAVE_TDM
    // K tile 64x128 bf16 via Tensor Data Mover (DMA overlaps V pair-packing)
    if (w == 0) {
      const unsigned short* gk =
          kbuf + ((size_t)(b * T_ + keyBase) * D_ + h * HD_);
      tdm_load_2d_bf16(ks_off, gk, (unsigned)D_, 1u << 20,
                       (unsigned long long)D_, (unsigned)HD_, 64u, 0u);
    }
#else
    {   // stage K tile 64x128 (u32 pairs along d)
      int key = tid >> 2;
      int dp0 = (tid & 3) * 16;
      size_t base = ((size_t)(b * T_ + keyBase + key) * D_ + h * HD_) >> 1;
      #pragma unroll
      for (int j = 0; j < 16; j++) Ks[key * 64 + dp0 + j] = k32[base + dp0 + j];
    }
#endif
    {   // stage V tile packed pairwise over keys
      int kp = tid >> 3;
      int n0 = (tid & 7) * 16;
      size_t b0 = ((size_t)(b * T_ + keyBase + 2 * kp) * D_ + h * HD_ + n0) >> 1;
      size_t b1 = b0 + (D_ >> 1);
      #pragma unroll
      for (int j = 0; j < 8; j++) {
        unsigned a = v32[b0 + j];
        unsigned c = v32[b1 + j];
        Vs[kp * 128 + n0 + 2 * j]     = (a & 0xFFFFu) | (c << 16);
        Vs[kp * 128 + n0 + 2 * j + 1] = (a >> 16) | (c & 0xFFFF0000u);
      }
    }
#if HAVE_TDM
    if (w == 0) __builtin_amdgcn_s_wait_tensorcnt(0);
#endif
    __syncthreads();
    if (keyBase > qr0 + 15) continue;   // wave-uniform: whole tile masked

    // S = Q K^T  (16x64 per wave)
    v8f s[4];
    #pragma unroll
    for (int j = 0; j < 4; j++) s[j] = zero8();
    #pragma unroll
    for (int j = 0; j < 4; j++) {
      int key = j * 16 + l16;
      #pragma unroll
      for (int ks = 0; ks < 4; ks++) {
        FragU bf;
        #pragma unroll
        for (int vv = 0; vv < 8; vv++)
          bf.u[vv] = Ks[key * 64 + ks * 16 + hl * 8 + vv];
        s[j] = __builtin_amdgcn_wmma_f32_16x16x32_bf16(
            false, qf[ks].v, false, bf.v, (short)0, s[j], false, false);
      }
    }

    // mask + ALiBi + online softmax (rows live in 16-lane halves)
    float fac[8];
    #pragma unroll
    for (int r = 0; r < 8; r++) {
      int qi = qr0 + r + 8 * hl;
      float mx = -3.0e38f;
      #pragma unroll
      for (int j = 0; j < 4; j++) {
        int kk = keyBase + j * 16 + l16;
        float sv = s[j][r] * scale + slope * (float)(kk - qi);
        if (kk > qi) sv = -3.0e38f;
        s[j][r] = sv;
        mx = fmaxf(mx, sv);
      }
      #pragma unroll
      for (int off = 1; off < 16; off <<= 1)
        mx = fmaxf(mx, __shfl_xor(mx, off, 32));
      float mnew = fmaxf(mrow[r], mx);
      fac[r] = __expf(mrow[r] - mnew);
      mrow[r] = mnew;
      float psum = 0.0f;
      #pragma unroll
      for (int j = 0; j < 4; j++) {
        float p = __expf(s[j][r] - mnew);
        psum += p;
        Ps[w][(r + 8 * hl) * 64 + j * 16 + l16] = f2bf(p);
      }
      #pragma unroll
      for (int off = 1; off < 16; off <<= 1)
        psum += __shfl_xor(psum, off, 32);
      lsum[r] = lsum[r] * fac[r] + psum;
      #pragma unroll
      for (int n = 0; n < 8; n++) o[n][r] *= fac[r];
    }
    __threadfence_block();   // order same-wave LDS P writes before fragment reads

    // O += P V  (P 16x64 as A-fragments from LDS)
    const unsigned* pu = (const unsigned*)Ps[w];
    #pragma unroll
    for (int ks2 = 0; ks2 < 2; ks2++) {
      FragU pf;
      #pragma unroll
      for (int vv = 0; vv < 8; vv++) {
        int kp2 = ks2 * 16 + hl * 8 + vv;
        pf.u[vv] = pu[l16 * 32 + kp2];
      }
      #pragma unroll
      for (int n = 0; n < 8; n++) {
        FragU vf;
        #pragma unroll
        for (int vv = 0; vv < 8; vv++)
          vf.u[vv] = Vs[(ks2 * 16 + hl * 8 + vv) * 128 + n * 16 + l16];
        o[n] = __builtin_amdgcn_wmma_f32_16x16x32_bf16(
            false, pf.v, false, vf.v, (short)0, o[n], false, false);
      }
    }
  }

  // normalize and store bf16 attention output (layout (B*T) x D, head columns)
  #pragma unroll
  for (int r = 0; r < 8; r++) {
    float inv = 1.0f / lsum[r];
    int row = qr0 + r + 8 * hl;
    size_t base = (size_t)(b * T_ + row) * D_ + h * HD_;
    #pragma unroll
    for (int n = 0; n < 8; n++)
      obuf[base + n * 16 + l16] = f2bf(o[n][r] * inv);
  }
}

extern "C" void kernel_launch(void* const* d_in, const int* in_sizes, int n_in,
                              void* d_out, int out_size, void* d_ws, size_t ws_size,
                              hipStream_t stream) {
  (void)in_sizes; (void)n_in; (void)out_size; (void)ws_size;
  const float* x      = (const float*)d_in[0];
  // d_in[1] = causal mask (bool) — computed analytically in-kernel
  const float* wq     = (const float*)d_in[2];
  const float* bq     = (const float*)d_in[3];
  const float* wk     = (const float*)d_in[4];
  const float* bk     = (const float*)d_in[5];
  const float* wv     = (const float*)d_in[6];
  const float* bv     = (const float*)d_in[7];
  const float* wo     = (const float*)d_in[8];
  const float* bo     = (const float*)d_in[9];
  const float* slopes = (const float*)d_in[10];

  const int M = B_ * T_;                       // 4096
  size_t elems = (size_t)M * D_;               // 8M elements per bf16 buffer
  unsigned short* qb = (unsigned short*)d_ws;
  unsigned short* kb = qb + elems;
  unsigned short* vb = kb + elems;
  unsigned short* ab = vb + elems;

  dim3 gg(D_ / 128, M / 128);                  // (16, 32)
  gemm_wmma<false, false><<<gg, 256, 0, stream>>>(x, wq, bq, qb, M, D_, D_);
  gemm_wmma<false, false><<<gg, 256, 0, stream>>>(x, wk, bk, kb, M, D_, D_);
  gemm_wmma<false, false><<<gg, 256, 0, stream>>>(x, wv, bv, vb, M, D_, D_);

  dim3 ga(T_ / 128, B_ * H_);                  // (16, 32)
  attn_wmma<<<ga, 256, 0, stream>>>(qb, kb, vb, slopes, ab);

  gemm_wmma<true, true><<<gg, 256, 0, stream>>>(ab, wo, bo, (float*)d_out, M, D_, D_);
}
